// Model_61478161875253
// MI455X (gfx1250) — 8-bit matrix operand order repaired and checked by reading, compiled, not run on hardware
//
#include <hip/hip_runtime.h>
#include <hip/hip_bf16.h>
#include <stdint.h>

#define BATCH    32
#define TSTEPS   250
#define IMGH     28
#define NCLASS   25
#define NUNITS   512
#define NZ       2048
#define CELLW    64
#define NHEADS   4
#define MEMN     250
#define FEATN    128
#define XPAD     160
#define KA       960
#define AOFF_RV  160
#define AOFF_H   416
#define PHID     320
#define KO       768
#define WGROWS   16
#define EPS_BN   1e-3f

typedef __attribute__((ext_vector_type(8))) int   v8i;
typedef __attribute__((ext_vector_type(8))) float v8f;

__device__ __forceinline__ float sigf(float x) { return 1.0f / (1.0f + __expf(-x)); }
__device__ __forceinline__ float tanf_(float x) {
  float e = __expf(2.0f * x);
  return (e - 1.0f) / (e + 1.0f);
}

__device__ __forceinline__ unsigned char f32_to_fp8(float f) {
  union { float f; unsigned u; } v; v.f = f;
  unsigned s = (v.u >> 24) & 0x80u;
  unsigned a = v.u & 0x7fffffffu;
  if (a >= 0x7f800000u) return (unsigned char)(s | 0x7f);
  if (a < 0x3c800000u) {
    float af = __builtin_fabsf(f);
    int d = (int)(af * 512.0f + 0.5f);
    return (unsigned char)(s | (unsigned)d);
  }
  unsigned r = a + 0x0007ffffu + ((a >> 20) & 1u);
  if (r >= 0x43e80000u) return (unsigned char)(s | 0x7e);
  unsigned e  = (r >> 23) - 120u;
  unsigned m  = (r >> 20) & 7u;
  return (unsigned char)(s | (e << 3) | m);
}

__device__ __forceinline__ v8i load_a8(const unsigned char* p) {
  v8i r;
#pragma unroll
  for (int j = 0; j < 4; ++j) {
    const unsigned int* q = (const unsigned int*)(p + j * 16);
    r[2 * j]     = (int)q[0];
    r[2 * j + 1] = (int)q[1];
  }
  return r;
}

__device__ __forceinline__ v8i load_b8(const unsigned char* p) {
  const unsigned int* q0 = (const unsigned int*)(p);
  const unsigned int* q1 = (const unsigned int*)(p + 32);
  v8i r;
  r[0] = (int)q0[0]; r[1] = (int)q0[1]; r[2] = (int)q0[4]; r[3] = (int)q0[5];
  r[4] = (int)q1[0]; r[5] = (int)q1[1]; r[6] = (int)q1[4]; r[7] = (int)q1[5];
  return r;
}

__global__ void repack_w1(const float* __restrict__ Wx, const float* __restrict__ Wh,
                          unsigned char* __restrict__ W1t) {
  int idx = blockIdx.x * 256 + threadIdx.x;
  if (idx >= NZ * KA) return;
  int n = idx / KA, k = idx % KA;
  float v = 0.0f;
  if (k < 153)                 v = Wx[k * NZ + n];
  else if (k >= 160 && k < 416) v = Wx[(k - 7) * NZ + n];
  else if (k >= 416 && k < 928) v = Wh[(k - 416) * NZ + n];
  W1t[idx] = f32_to_fp8(v);
}

__global__ void repack_wp(const float* __restrict__ Wp, unsigned char* __restrict__ WpT) {
  int idx = blockIdx.x * 256 + threadIdx.x;
  if (idx >= PHID * NUNITS) return;
  int n = idx / NUNITS, k = idx % NUNITS;
  WpT[idx] = f32_to_fp8(Wp[k * PHID + n]);
}

__global__ void repack_wo(const float* __restrict__ Wo, unsigned char* __restrict__ WoT) {
  int idx = blockIdx.x * 256 + threadIdx.x;
  if (idx >= NUNITS * KO) return;
  int n = idx / KO, k = idx % KO;
  float v = (k < 256) ? Wo[(512 + k) * NUNITS + n] : Wo[(k - 256) * NUNITS + n];
  WoT[idx] = f32_to_fp8(v);
}

__global__ void repack_wf(const float* __restrict__ Wf, unsigned char* __restrict__ WfT) {
  int idx = blockIdx.x * 256 + threadIdx.x;
  if (idx >= 32 * NUNITS) return;
  int n = idx / NUNITS, k = idx % NUNITS;
  float v = (n < NCLASS) ? Wf[k * NCLASS + n] : 0.0f;
  WfT[idx] = f32_to_fp8(v);
}

__global__ __launch_bounds__(128) void conv_kernel(
    const float* __restrict__ images, const int* __restrict__ labels,
    const float* __restrict__ k1, const float* __restrict__ cb1, const float* __restrict__ g1,
    const float* __restrict__ be1, const float* __restrict__ mm1, const float* __restrict__ mv1,
    const float* __restrict__ k2, const float* __restrict__ cb2, const float* __restrict__ g2,
    const float* __restrict__ be2, const float* __restrict__ mm2, const float* __restrict__ mv2,
    const float* __restrict__ k3, const float* __restrict__ cb3, const float* __restrict__ g3,
    const float* __restrict__ be3, const float* __restrict__ mm3, const float* __restrict__ mv3,
    unsigned char* __restrict__ Xg) {
  __shared__ float img[IMGH * IMGH];
  __shared__ float a1[13 * 13 * 8];
  __shared__ float a2[6 * 6 * 16];
  const int id  = blockIdx.x;
  const int tid = threadIdx.x;
  const float* im = images + (size_t)id * (IMGH * IMGH);
  for (int i = tid; i < IMGH * IMGH; i += 128) img[i] = im[i];
  __syncthreads();
  for (int i = tid; i < 13 * 13 * 8; i += 128) {
    int oc = i & 7, tt = i >> 3, ox = tt % 13, oy = tt / 13;
    float acc = cb1[oc];
#pragma unroll
    for (int ky = 0; ky < 3; ++ky)
#pragma unroll
      for (int kx = 0; kx < 3; ++kx)
        acc += img[(oy * 2 + ky) * IMGH + ox * 2 + kx] * k1[(ky * 3 + kx) * 8 + oc];
    float y = (acc - mm1[oc]) * g1[oc] * rsqrtf(mv1[oc] + EPS_BN) + be1[oc];
    a1[i] = fmaxf(y, 0.0f);
  }
  __syncthreads();
  for (int i = tid; i < 6 * 6 * 16; i += 128) {
    int oc = i & 15, tt = i >> 4, ox = tt % 6, oy = tt / 6;
    float acc = cb2[oc];
    for (int ky = 0; ky < 3; ++ky)
      for (int kx = 0; kx < 3; ++kx)
#pragma unroll
        for (int ic = 0; ic < 8; ++ic)
          acc += a1[((oy * 2 + ky) * 13 + ox * 2 + kx) * 8 + ic] * k2[((ky * 3 + kx) * 8 + ic) * 16 + oc];
    float y = (acc - mm2[oc]) * g2[oc] * rsqrtf(mv2[oc] + EPS_BN) + be2[oc];
    a2[i] = fmaxf(y, 0.0f);
  }
  __syncthreads();
  unsigned char* xrow = Xg + (size_t)id * XPAD;
  for (int i = tid; i < FEATN; i += 128) {
    int oc = i & 31, tt = i >> 5, ox = tt & 1, oy = tt >> 1;
    float acc = cb3[oc];
    for (int ky = 0; ky < 3; ++ky)
      for (int kx = 0; kx < 3; ++kx)
#pragma unroll
        for (int ic = 0; ic < 16; ++ic)
          acc += a2[((oy * 2 + ky) * 6 + ox * 2 + kx) * 16 + ic] * k3[((ky * 3 + kx) * 16 + ic) * 32 + oc];
    float y = (acc - mm3[oc]) * g3[oc] * rsqrtf(mv3[oc] + EPS_BN) + be3[oc];
    xrow[i] = f32_to_fp8(fmaxf(y, 0.0f));
  }
  int lab = labels[id];
  for (int j = tid; j < 32; j += 128) {
    float v = (j < NCLASS && lab == j) ? 1.0f : 0.0f;
    xrow[FEATN + j] = f32_to_fp8(v);
  }
}

#define SMEM_BYTES (WGROWS * KA + WGROWS * NZ * 4 + WGROWS * NUNITS * 4 + WGROWS * NUNITS)

__global__ __launch_bounds__(256) void lstm_kernel(
    const unsigned char* __restrict__ Xg,
    const unsigned char* __restrict__ W1t, const float* __restrict__ bl,
    const unsigned char* __restrict__ WpT, const float* __restrict__ bp,
    const unsigned char* __restrict__ WoT, const float* __restrict__ bo,
    const unsigned char* __restrict__ WfT, const float* __restrict__ bf,
    float* __restrict__ memg, float* __restrict__ rnormg,
    float* __restrict__ outp) {
  extern __shared__ char smem_raw[];
  unsigned char* sA8 = (unsigned char*)smem_raw;
  float* sz  = (float*)(smem_raw + WGROWS * KA);
  float* sc  = sz + WGROWS * NZ;
  unsigned char* sAo = (unsigned char*)(sc + WGROWS * NUNITS);
  __shared__ float keynorm[WGROWS * NHEADS];

  const int tid  = threadIdx.x;
  const int lane = tid & 31;
  const int wave = tid >> 5;
  const int bg   = blockIdx.x;
  const int row  = lane & 15;
  const int abase = (lane & 16) ? 8 : 0;
  const int bbase = (lane & 16) ? 8 : 0;
  const int mb    = (lane & 16) ? 8 : 0;

  float* memb = memg + (size_t)bg * WGROWS * MEMN * CELLW;
  float* rnb  = rnormg + bg * WGROWS * MEMN;

  for (int i = tid; i < WGROWS * KA; i += 256) sA8[i] = 0;
  for (int i = tid; i < WGROWS * NUNITS; i += 256) sc[i] = 0.0f;
  for (int i = tid; i < WGROWS * MEMN * CELLW; i += 256) memb[i] = 0.0f;
  for (int i = tid; i < WGROWS * MEMN; i += 256) rnb[i] = 1.0f;
  __syncthreads();

  for (int t = 0; t < TSTEPS; ++t) {
    for (int i = tid; i < WGROWS * XPAD; i += 256) {
      int m = i / XPAD, k = i % XPAD;
      sA8[m * KA + k] = Xg[((size_t)(bg * WGROWS + m) * TSTEPS + t) * XPAD + k];
    }
    if (t + 1 < TSTEPS) {
      int i = tid;
      if (i < WGROWS * 2) {
        int m = i >> 1, half = i & 1;
        __builtin_prefetch(Xg + ((size_t)(bg * WGROWS + m) * TSTEPS + (t + 1)) * XPAD + half * 128, 0, 1);
      }
    }
    __syncthreads();

    {
      const unsigned char* arow = sA8 + row * KA + abase;
      v8i afr[KA / 64];
#pragma unroll
      for (int kt = 0; kt < KA / 64; ++kt) afr[kt] = load_a8(arow + kt * 64);
      for (int nt = wave; nt < NZ / 16; nt += 8) {
        v8f acc = {};
        const unsigned char* brow = W1t + (size_t)(nt * 16 + row) * KA + bbase;
#pragma unroll
        for (int kt = 0; kt < KA / 64; ++kt) {
          v8i b = load_b8(brow + kt * 64);
          acc = __builtin_amdgcn_wmma_f32_16x16x64_fp8_fp8(afr[kt], b, (short)0, acc, false, false);
        }
        int n = nt * 16 + row;
#pragma unroll
        for (int r2 = 0; r2 < 8; ++r2) sz[(mb + r2) * NZ + n] = acc[r2];
      }
    }
    __syncthreads();

    for (int i = tid; i < WGROWS * NUNITS; i += 256) {
      int m = i >> 9, u = i & 511;
      const float* zr = sz + m * NZ;
      float zi = zr[u] + bl[u];
      float zf = zr[512 + u] + bl[512 + u];
      float zg = zr[1024 + u] + bl[1024 + u];
      float zo = zr[1536 + u] + bl[1536 + u];
      float ig = sigf(zi), fg = sigf(zf), gg = tanf_(zg), og = sigf(zo);
      float c = fg * sc[i] + ig * gg;
      sc[i] = c;
      float h = og * tanf_(c);
      sA8[m * KA + AOFF_H + u] = f32_to_fp8(h);
    }
    __syncthreads();

    float* hidb = sz;
    float* attb = sz + WGROWS * PHID;
    {
      const unsigned char* arow = sA8 + row * KA + AOFF_H + abase;
      v8i afr[NUNITS / 64];
#pragma unroll
      for (int kt = 0; kt < NUNITS / 64; ++kt) afr[kt] = load_a8(arow + kt * 64);
      for (int nt = wave; nt < PHID / 16; nt += 8) {
        v8f acc = {};
        const unsigned char* brow = WpT + (size_t)(nt * 16 + row) * NUNITS + bbase;
#pragma unroll
        for (int kt = 0; kt < NUNITS / 64; ++kt) {
          v8i b = load_b8(brow + kt * 64);
          acc = __builtin_amdgcn_wmma_f32_16x16x64_fp8_fp8(afr[kt], b, (short)0, acc, false, false);
        }
        int n = nt * 16 + row;
#pragma unroll
        for (int r2 = 0; r2 < 8; ++r2) hidb[(mb + r2) * PHID + n] = acc[r2] + bp[n];
      }
    }
    __syncthreads();

    for (int i = tid; i < WGROWS * PHID; i += 256) hidb[i] = tanf_(hidb[i]);
    __syncthreads();

    if (tid < WGROWS * NHEADS) {
      int m = tid >> 2, r = tid & 3;
      const float* kp = hidb + m * PHID + CELLW + r * CELLW;
      float s = 0.0f;
#pragma unroll 8
      for (int j = 0; j < CELLW; ++j) s += kp[j] * kp[j];
      keynorm[tid] = rsqrtf(fmaxf(s, 1e-12f));
    }
    __syncthreads();

    for (int i = tid; i < WGROWS * MEMN; i += 256) {
      int m2 = i / MEMN, m = i % MEMN;
      const float* mp = memb + ((size_t)m2 * MEMN + m) * CELLW;
      const float* kp = hidb + m2 * PHID + CELLW;
      float d0 = 0, d1 = 0, d2 = 0, d3 = 0;
#pragma unroll 4
      for (int j = 0; j < CELLW; ++j) {
        float mv = mp[j];
        d0 += kp[j] * mv;
        d1 += kp[CELLW + j] * mv;
        d2 += kp[2 * CELLW + j] * mv;
        d3 += kp[3 * CELLW + j] * mv;
      }
      float rn = rnb[m2 * MEMN + m];
      attb[(m2 * NHEADS + 0) * MEMN + m] = d0 * keynorm[m2 * 4 + 0] * rn;
      attb[(m2 * NHEADS + 1) * MEMN + m] = d1 * keynorm[m2 * 4 + 1] * rn;
      attb[(m2 * NHEADS + 2) * MEMN + m] = d2 * keynorm[m2 * 4 + 2] * rn;
      attb[(m2 * NHEADS + 3) * MEMN + m] = d3 * keynorm[m2 * 4 + 3] * rn;
    }
    __syncthreads();

    if (tid < WGROWS * NHEADS) {
      float* ar = attb + tid * MEMN;
      float mx = ar[0];
      for (int m = 1; m < MEMN; ++m) mx = fmaxf(mx, ar[m]);
      float s = 0.0f;
      for (int m = 0; m < MEMN; ++m) { float e = __expf(ar[m] - mx); ar[m] = e; s += e; }
      float inv = 1.0f / s;
      for (int m = 0; m < MEMN; ++m) ar[m] *= inv;
    }
    __syncthreads();

    for (int i = tid; i < WGROWS * CELLW; i += 256) {
      int m2 = i >> 6, s = i & 63;
      const float* mp = memb + (size_t)m2 * MEMN * CELLW + s;
      const float* w0 = attb + (m2 * NHEADS + 0) * MEMN;
      const float* w1 = attb + (m2 * NHEADS + 1) * MEMN;
      const float* w2 = attb + (m2 * NHEADS + 2) * MEMN;
      const float* w3 = attb + (m2 * NHEADS + 3) * MEMN;
      float a0 = 0, a1 = 0, a2 = 0, a3 = 0;
      for (int m = 0; m < MEMN; ++m) {
        float mv = mp[(size_t)m * CELLW];
        a0 += w0[m] * mv; a1 += w1[m] * mv; a2 += w2[m] * mv; a3 += w3[m] * mv;
      }
      unsigned char* adst = sA8 + m2 * KA + AOFF_RV;
      adst[0 * CELLW + s] = f32_to_fp8(a0);
      adst[1 * CELLW + s] = f32_to_fp8(a1);
      adst[2 * CELLW + s] = f32_to_fp8(a2);
      adst[3 * CELLW + s] = f32_to_fp8(a3);
    }
    __syncthreads();

    {
      int slot = t % MEMN;
      for (int i = tid; i < WGROWS * CELLW; i += 256) {
        int m2 = i >> 6, s = i & 63;
        memb[((size_t)m2 * MEMN + slot) * CELLW + s] = hidb[m2 * PHID + s];
      }
      if (tid < WGROWS) {
        const float* wv = hidb + tid * PHID;
        float s = 0.0f;
#pragma unroll 8
        for (int j = 0; j < CELLW; ++j) s += wv[j] * wv[j];
        rnb[tid * MEMN + slot] = rsqrtf(fmaxf(s, 1e-12f));
      }
    }
    __syncthreads();

    {
      const unsigned char* arow = sA8 + row * KA + AOFF_RV + abase;
      v8i afr[KO / 64];
#pragma unroll
      for (int kt = 0; kt < KO / 64; ++kt) afr[kt] = load_a8(arow + kt * 64);
      for (int nt = wave; nt < NUNITS / 16; nt += 8) {
        v8f acc = {};
        const unsigned char* brow = WoT + (size_t)(nt * 16 + row) * KO + bbase;
#pragma unroll
        for (int kt = 0; kt < KO / 64; ++kt) {
          v8i b = load_b8(brow + kt * 64);
          acc = __builtin_amdgcn_wmma_f32_16x16x64_fp8_fp8(afr[kt], b, (short)0, acc, false, false);
        }
        int n = nt * 16 + row;
#pragma unroll
        for (int r2 = 0; r2 < 8; ++r2) sz[(mb + r2) * NUNITS + n] = acc[r2] + bo[n];
      }
    }
    __syncthreads();

    for (int i = tid; i < WGROWS * NUNITS; i += 256) sAo[i] = f32_to_fp8(tanf_(sz[i]));
    __syncthreads();

    float* lbuf = sz + WGROWS * NUNITS;
    if (wave < 2) {
      int nt = wave;
      v8f acc = {};
      const unsigned char* arow = sAo + row * NUNITS + abase;
      const unsigned char* brow = WfT + (size_t)(nt * 16 + row) * NUNITS + bbase;
#pragma unroll
      for (int kt = 0; kt < NUNITS / 64; ++kt) {
        v8i a = load_a8(arow + kt * 64);
        v8i b = load_b8(brow + kt * 64);
        acc = __builtin_amdgcn_wmma_f32_16x16x64_fp8_fp8(a, b, (short)0, acc, false, false);
      }
      int n = nt * 16 + row;
#pragma unroll
      for (int r2 = 0; r2 < 8; ++r2)
        lbuf[(mb + r2) * 32 + n] = acc[r2] + ((n < NCLASS) ? bf[n] : 0.0f);
    }
    __syncthreads();

    if (tid < WGROWS) {
      const float* lr = lbuf + tid * 32;
      float mx = lr[0];
      for (int j = 1; j < NCLASS; ++j) mx = fmaxf(mx, lr[j]);
      float s = 0.0f;
      for (int j = 0; j < NCLASS; ++j) s += __expf(lr[j] - mx);
      float inv = 1.0f / s;
      float* op = outp + ((size_t)(bg * WGROWS + tid) * TSTEPS + t) * NCLASS;
      for (int j = 0; j < NCLASS; ++j) op[j] = __expf(lr[j] - mx) * inv;
    }
    __syncthreads();
  }
}

extern "C" void kernel_launch(void* const* d_in, const int* in_sizes, int n_in,
                              void* d_out, int out_size, void* d_ws, size_t ws_size,
                              hipStream_t stream) {
  (void)in_sizes; (void)n_in; (void)out_size; (void)ws_size;
  const float* images = (const float*)d_in[0];
  const int*   labels = (const int*)d_in[1];
  const float* k1 = (const float*)d_in[2],  *cb1 = (const float*)d_in[3],
             * g1 = (const float*)d_in[4],  *be1 = (const float*)d_in[5],
             * mm1 = (const float*)d_in[6], *mv1 = (const float*)d_in[7];
  const float* k2 = (const float*)d_in[8],  *cb2 = (const float*)d_in[9],
             * g2 = (const float*)d_in[10], *be2 = (const float*)d_in[11],
             * mm2 = (const float*)d_in[12],*mv2 = (const float*)d_in[13];
  const float* k3 = (const float*)d_in[14], *cb3 = (const float*)d_in[15],
             * g3 = (const float*)d_in[16], *be3 = (const float*)d_in[17],
             * mm3 = (const float*)d_in[18],*mv3 = (const float*)d_in[19];
  const float* Wx = (const float*)d_in[20];
  const float* Wh = (const float*)d_in[21];
  const float* bl = (const float*)d_in[22];
  const float* Wp = (const float*)d_in[23];
  const float* bp = (const float*)d_in[24];
  const float* Wo = (const float*)d_in[25];
  const float* bo = (const float*)d_in[26];
  const float* Wf = (const float*)d_in[27];
  const float* bf = (const float*)d_in[28];

  char* ws = (char*)d_ws;
  auto carve = [&](size_t bytes) -> char* {
    char* p = ws;
    ws += (bytes + 255) & ~(size_t)255;
    return p;
  };
  unsigned char* Xg  = (unsigned char*)carve((size_t)BATCH * TSTEPS * XPAD);
  unsigned char* W1t = (unsigned char*)carve((size_t)NZ * KA);
  unsigned char* WpT = (unsigned char*)carve((size_t)PHID * NUNITS);
  unsigned char* WoT = (unsigned char*)carve((size_t)NUNITS * KO);
  unsigned char* WfT = (unsigned char*)carve((size_t)32 * NUNITS);
  float* memg   = (float*)carve((size_t)BATCH * MEMN * CELLW * 4);
  float* rnormg = (float*)carve((size_t)BATCH * MEMN * 4);

  repack_w1<<<(NZ * KA + 255) / 256, 256, 0, stream>>>(Wx, Wh, W1t);
  repack_wp<<<(PHID * NUNITS + 255) / 256, 256, 0, stream>>>(Wp, WpT);
  repack_wo<<<(NUNITS * KO + 255) / 256, 256, 0, stream>>>(Wo, WoT);
  repack_wf<<<(32 * NUNITS + 255) / 256, 256, 0, stream>>>(Wf, WfT);

  conv_kernel<<<BATCH * TSTEPS, 128, 0, stream>>>(
      images, labels,
      k1, cb1, g1, be1, mm1, mv1,
      k2, cb2, g2, be2, mm2, mv2,
      k3, cb3, g3, be3, mm3, mv3,
      Xg);

  lstm_kernel<<<BATCH / WGROWS, 256, SMEM_BYTES, stream>>>(
      Xg, W1t, bl, WpT, bp, WoT, bo, WfT, bf, memg, rnormg, (float*)d_out);
}
